// RecurrentGCN_58085137711403
// MI455X (gfx1250) — matrix operand order repaired and checked by reading, not compiled, not run on hardware
//
#include <hip/hip_runtime.h>
#include <hip/hip_bf16.h>

// ---------------------------------------------------------------------------
// HeteroGCLSTM forward with zero initial state collapses to:
//   ga[g] = X @ W[g] + b[g] + conv_bl[g][0],  g in {i, c, o}   (forget gate dead)
//   out   = relu( sig(ga_o)*tanh( sig(ga_i)*tanh(ga_c) ) ) @ lin_W + lin_b
// One (50000x768)@(768x192) GEMM -> HBM-bound (154 MB @ 23.3 TB/s ~ 6.6us).
// f16 WMMA (v_wmma_f32_16x16x32_f16) with f32 accum; X read once, weights in L2.
// ---------------------------------------------------------------------------

typedef __attribute__((ext_vector_type(16))) _Float16 v16h;
typedef __attribute__((ext_vector_type(8)))  float    v8f;

#define N_ART   50000
#define F_ART   768
#define HID     64
#define M_TILE  128
#define N_TILES 12          // 3 live gates * 64 cols / 16
#define KT      32
#define XSTR    34          // 17 dwords: odd stride -> conflict-free on 64 banks
#define WSTR    34

union Frag16 { v16h h; unsigned int u[8]; };
union Acc    { v8f  f; float s[8]; };

__global__ __launch_bounds__(256)
void hetero_lstm_fused(const float* __restrict__ X,     // (50000, 768)
                       const float* __restrict__ Wg,    // (4, 768, 64)
                       const float* __restrict__ ba,    // (4, 1, 64)
                       const float* __restrict__ cbl,   // (4, 3, 64)
                       const float* __restrict__ linW,  // (64, 2)
                       const float* __restrict__ linb,  // (2,)
                       float* __restrict__ out)         // (50000, 2)
{
    __shared__ _Float16 Xl[M_TILE * XSTR];       // X tile, f16, [m][k]
    __shared__ _Float16 Wl[(N_TILES * 16) * WSTR]; // W tile, f16, [n][k] (transposed)

    const int tid  = threadIdx.x;
    const int wave = tid >> 5;       // 0..7, wave32
    const int lane = tid & 31;
    const int ln   = lane & 15;      // position within 16-lane half
    const int lg   = lane >> 4;      // half-wave id
    const int blockRow = blockIdx.x * M_TILE;

    Acc acc[N_TILES];
#pragma unroll
    for (int i = 0; i < N_TILES; ++i)
#pragma unroll
        for (int j = 0; j < 8; ++j) acc[i].s[j] = 0.0f;

    for (int kb = 0; kb < F_ART; kb += KT) {
        // ---- stage X tile: 128 rows x 32 k, f32 -> f16 ---------------------
#pragma unroll
        for (int i = 0; i < 4; ++i) {
            int fi = tid + i * 256;          // 0..1023 float4 slots
            int r  = fi >> 3;                // 0..127
            int cq = fi & 7;                 // float4 within row
            int gr = blockRow + r;
            float4 v = make_float4(0.f, 0.f, 0.f, 0.f);
            if (gr < N_ART) {
                v = *(const float4*)(X + (size_t)gr * F_ART + kb + cq * 4);
                if (kb + KT < F_ART)  // pull next K-tile toward the caches
                    __builtin_prefetch(X + (size_t)gr * F_ART + kb + KT + cq * 4, 0, 3);
            }
            _Float16* dst = &Xl[r * XSTR + cq * 4];
            dst[0] = (_Float16)v.x; dst[1] = (_Float16)v.y;
            dst[2] = (_Float16)v.z; dst[3] = (_Float16)v.w;
        }
        // ---- stage W tile: gates {0,2,3}, 32 k x 64 h, transposed ----------
#pragma unroll
        for (int i = 0; i < 24; ++i) {
            int fi  = tid + i * 256;         // 0..6143
            int h   = fi & 63;               // coalesced over h
            int row = fi >> 6;               // 0..95
            int gb  = row >> 5;              // gate block 0..2
            int kk  = row & 31;
            int g   = gb ? gb + 1 : 0;       // {0,2,3} = i, cand, o
            float v = Wg[((size_t)g * F_ART + kb + kk) * HID + h];
            Wl[(gb * 64 + h) * WSTR + kk] = (_Float16)v;
        }
        __syncthreads();

        // ---- A fragment (16x32 f16): ISA 7.12.2 layout ---------------------
        Frag16 A;
#pragma unroll
        for (int v = 0; v < 8; ++v) {
            int k0 = (v < 4 ? 2 * v : 16 + 2 * (v - 4)) + lg * 8;
            A.u[v] = *(const unsigned int*)&Xl[(wave * 16 + ln) * XSTR + k0];
        }
        // ---- 12 WMMAs over the n-tiles ------------------------------------
#pragma unroll
        for (int nt = 0; nt < N_TILES; ++nt) {
            Frag16 B;
#pragma unroll
            for (int v = 0; v < 8; ++v) {
                int k0 = (v < 4 ? 2 * v : 16 + 2 * (v - 4)) + lg * 8;    // B: same k order as A (lanes 0-15 K=0..7,16..23; 16-31 K=8..15,24..31)
                B.u[v] = *(const unsigned int*)&Wl[(nt * 16 + ln) * WSTR + k0];
            }
            acc[nt].f = __builtin_amdgcn_wmma_f32_16x16x32_f16(
                false, A.h, false, B.h, (short)0, acc[nt].f, false, false);
        }
        __syncthreads();
    }

    // ---- epilogue: bias + LSTM gating + 64->2 projection -------------------
    // For fixed (lane, v): gate i in acc[nt], cand in acc[nt+4], o in acc[nt+8],
    // all at the same (m = v + 8*lg, h = nt*16 + ln).
    float bi[4], bt[4], bo[4], lw0[4], lw1[4];
#pragma unroll
    for (int nt = 0; nt < 4; ++nt) {
        int h  = nt * 16 + ln;
        bi[nt] = ba[0 * HID + h] + cbl[(0 * 3 + 0) * HID + h];
        bt[nt] = ba[2 * HID + h] + cbl[(2 * 3 + 0) * HID + h];
        bo[nt] = ba[3 * HID + h] + cbl[(3 * 3 + 0) * HID + h];
        lw0[nt] = linW[h * 2 + 0];
        lw1[nt] = linW[h * 2 + 1];
    }
    const float lb0 = linb[0], lb1 = linb[1];

#pragma unroll
    for (int v = 0; v < 8; ++v) {
        float p0 = 0.f, p1 = 0.f;
#pragma unroll
        for (int nt = 0; nt < 4; ++nt) {
            float gi = acc[nt    ].s[v] + bi[nt];
            float gt = acc[nt + 4].s[v] + bt[nt];
            float go = acc[nt + 8].s[v] + bo[nt];
            float ii = 1.f / (1.f + __expf(-gi));
            float oo = 1.f / (1.f + __expf(-go));
            float c  = ii * tanhf(gt);
            float hn = oo * tanhf(c);
            float r  = fmaxf(hn, 0.f);
            p0 += r * lw0[nt];
            p1 += r * lw1[nt];
        }
        // reduce over the 16 lanes of this half-wave (same m, different h)
#pragma unroll
        for (int off = 8; off >= 1; off >>= 1) {
            p0 += __shfl_xor(p0, off, 32);
            p1 += __shfl_xor(p1, off, 32);
        }
        int row = blockRow + wave * 16 + (v + 8 * lg);
        if (ln == 0 && row < N_ART) {
            out[row * 2 + 0] = p0 + lb0;
            out[row * 2 + 1] = p1 + lb1;
        }
    }
}

extern "C" void kernel_launch(void* const* d_in, const int* in_sizes, int n_in,
                              void* d_out, int out_size, void* d_ws, size_t ws_size,
                              hipStream_t stream) {
    (void)in_sizes; (void)n_in; (void)out_size; (void)d_ws; (void)ws_size;
    const float* x_article = (const float*)d_in[0];   // (50000, 768)
    // d_in[1..4]: x_user + all edge lists are dead (h0 = c0 = 0)
    const float* W_article = (const float*)d_in[5];   // (4, 768, 64)
    const float* b_article = (const float*)d_in[7];   // (4, 1, 64)
    const float* conv_bl   = (const float*)d_in[10];  // (4, 3, 64)
    const float* lin_W     = (const float*)d_in[12];  // (64, 2)
    const float* lin_b     = (const float*)d_in[13];  // (2,)
    float* out = (float*)d_out;

    dim3 grid((N_ART + M_TILE - 1) / M_TILE);         // 391 workgroups
    dim3 block(256);                                  // 8 wave32 waves
    hetero_lstm_fused<<<grid, block, 0, stream>>>(
        x_article, W_article, b_article, conv_bl, lin_W, lin_b, out);
}
